// FlowModule_37915971289890
// MI455X (gfx1250) — matrix operand order repaired and checked by reading, not compiled, not run on hardware
//
#include <hip/hip_runtime.h>
#include <hip/hip_bf16.h>

// ---------------- problem constants ----------------
#define BATCH 8
#define H     128
#define W     128
#define C     128
#define NF    128
#define MD    4
#define HP    (H + 2*MD)     // 136
#define WP    (W + 2*MD)     // 136
#define CIN   213            // 2 + 2 + 81 + 128
#define K0P   224            // CIN padded up to multiple of 32 for WMMA K loop
#define M_TOT (BATCH*H*W)    // 131072 pixels

typedef __attribute__((ext_vector_type(16))) __bf16       v16bf;
typedef __attribute__((ext_vector_type(8)))  float        v8f;
typedef __attribute__((ext_vector_type(4)))  unsigned int v4u;
typedef __attribute__((ext_vector_type(4)))  __bf16       v4bf;
typedef __attribute__((ext_vector_type(4)))  unsigned int u32x4;
typedef __attribute__((ext_vector_type(8)))  int          i32x8;
typedef __attribute__((ext_vector_type(4)))  int          i32x4;

union FragU { v4u u2[2]; v16bf v; };  // 32 bytes: one WMMA 16-bit A/B fragment

// ---------------------------------------------------------------------------
// Kernel 1: conv-transpose upsample of features (128->2) and flow (2->2),
// bilinear mask resize + sigmoid gate. Writes bf16 concat channels [0..4),
// fp32 up_flow into d_out, fp32 gate scratch.
// lax.conv_transpose(k=4,s=2,'SAME') => taps kh with kh%2==oy%2, iy=(oy+kh-2)/2
// ---------------------------------------------------------------------------
__global__ void upsample_kernel(const float* __restrict__ sf,
                                const float* __restrict__ sfl,
                                const float* __restrict__ sm,
                                const float* __restrict__ upf_k,
                                const float* __restrict__ upf_b,
                                const float* __restrict__ upfl_k,
                                const float* __restrict__ upfl_b,
                                __bf16* __restrict__ cat,
                                float* __restrict__ upflow_out,
                                float* __restrict__ gate)
{
    int p = blockIdx.x * blockDim.x + threadIdx.x;
    if (p >= M_TOT) return;
    int b  = p / (H*W);
    int rm = p - b*(H*W);
    int oy = rm / W;
    int ox = rm - oy*W;
    const int HS = H/2, WS = W/2;

    float f0 = 0.f, f1 = 0.f, fl0 = 0.f, fl1 = 0.f;
    int khs = (oy & 1) ? 1 : 0;
    int kws = (ox & 1) ? 1 : 0;
    for (int kh = khs; kh < 4; kh += 2) {
        int dy2 = oy + kh - 2;
        if (dy2 < 0) continue;
        int iy = dy2 >> 1;
        if (iy >= HS) continue;
        for (int kw = kws; kw < 4; kw += 2) {
            int dx2 = ox + kw - 2;
            if (dx2 < 0) continue;
            int ix = dx2 >> 1;
            if (ix >= WS) continue;
            const float* srcf = sf + ((size_t)(b*HS + iy)*WS + ix)*NF;
            const float* kf   = upf_k + (size_t)(kh*4 + kw)*NF*2;
            float a0 = 0.f, a1 = 0.f;
            for (int ic = 0; ic < NF; ic += 2) {
                float v0 = srcf[ic], v1 = srcf[ic+1];
                a0 += v0*kf[ic*2+0] + v1*kf[ic*2+2];
                a1 += v0*kf[ic*2+1] + v1*kf[ic*2+3];
            }
            f0 += a0; f1 += a1;
            const float* srcl = sfl + ((size_t)(b*HS + iy)*WS + ix)*2;
            const float* kl   = upfl_k + (kh*4 + kw)*4;
            fl0 += srcl[0]*kl[0] + srcl[1]*kl[2];
            fl1 += srcl[0]*kl[1] + srcl[1]*kl[3];
        }
    }
    f0 += upf_b[0]; f1 += upf_b[1];
    float w0 = (fl0 + upfl_b[0]) * 2.0f;   // FlowUpsample *2
    float w1 = (fl1 + upfl_b[1]) * 2.0f;

    // bilinear mask resize 64->128, half-pixel centers, edge clamp
    float sy = fminf(fmaxf(oy*0.5f - 0.25f, 0.f), (float)(HS-1));
    float sx = fminf(fmaxf(ox*0.5f - 0.25f, 0.f), (float)(WS-1));
    int y0 = min((int)sy, HS-2);
    int x0 = min((int)sx, WS-2);
    float ay = sy - (float)y0, ax = sx - (float)x0;
    const float* mb = sm + (size_t)b*HS*WS;
    float m00 = mb[y0*WS + x0],     m01 = mb[y0*WS + x0 + 1];
    float m10 = mb[(y0+1)*WS + x0], m11 = mb[(y0+1)*WS + x0 + 1];
    float top = m00 + ax*(m01 - m00);
    float bot = m10 + ax*(m11 - m10);
    float mval = top + ay*(bot - top);
    float g = 1.0f / (1.0f + __expf(-mval));

    __bf16* cr = cat + (size_t)p*K0P;
    cr[0] = (__bf16)f0; cr[1] = (__bf16)f1;
    cr[2] = (__bf16)w0; cr[3] = (__bf16)w1;
    upflow_out[p*2 + 0] = w0;
    upflow_out[p*2 + 1] = w1;
    gate[p] = g;
}

// ---------------------------------------------------------------------------
// Kernel 2: warp(current, up_flow) * sigmoid(mask)  -> zero-padded fp32 buffer
// tfa interpolate_bilinear semantics. Thread per (pixel, 4 channels).
// ---------------------------------------------------------------------------
__global__ void warp_kernel(const float* __restrict__ cur,
                            const float* __restrict__ upflow,
                            const float* __restrict__ gate,
                            float* __restrict__ pad)
{
    int gid = blockIdx.x * blockDim.x + threadIdx.x;
    int p = gid >> 5;            // 32 channel groups of 4
    int c = (gid & 31) * 4;
    if (p >= M_TOT) return;
    int b  = p / (H*W);
    int rm = p - b*(H*W);
    int y  = rm / W;
    int x  = rm - y*W;

    float qy = (float)y + upflow[p*2 + 0] * (1.0f/20.0f);
    float qx = (float)x + upflow[p*2 + 1] * (1.0f/20.0f);
    float y0f = fminf(fmaxf(floorf(qy), 0.f), (float)(H-2));
    float x0f = fminf(fmaxf(floorf(qx), 0.f), (float)(W-2));
    float ay = fminf(fmaxf(qy - y0f, 0.f), 1.f);
    float ax = fminf(fmaxf(qx - x0f, 0.f), 1.f);
    int y0 = (int)y0f, x0 = (int)x0f;

    const float* base = cur + ((size_t)(b*H + y0)*W + x0)*C + c;
    float4 tl = *(const float4*)(base);
    float4 tr = *(const float4*)(base + C);
    float4 bl = *(const float4*)(base + (size_t)W*C);
    float4 br = *(const float4*)(base + (size_t)W*C + C);
    float g = gate[p];
    float4 o;
    {
        float t = tl.x + ax*(tr.x - tl.x), bo = bl.x + ax*(br.x - bl.x);
        o.x = (t + ay*(bo - t)) * g;
    }
    {
        float t = tl.y + ax*(tr.y - tl.y), bo = bl.y + ax*(br.y - bl.y);
        o.y = (t + ay*(bo - t)) * g;
    }
    {
        float t = tl.z + ax*(tr.z - tl.z), bo = bl.z + ax*(br.z - bl.z);
        o.z = (t + ay*(bo - t)) * g;
    }
    {
        float t = tl.w + ax*(tr.w - tl.w), bo = bl.w + ax*(br.w - bl.w);
        o.w = (t + ay*(bo - t)) * g;
    }
    *(float4*)(pad + (((size_t)(b*HP + (y+MD))*WP + (x+MD))*C + c)) = o;
}

// ---------------------------------------------------------------------------
// Kernel 3: copy prev (fp32) into concat channels [85..213) as bf16
// ---------------------------------------------------------------------------
__global__ void prevcat_kernel(const float* __restrict__ prev,
                               __bf16* __restrict__ cat)
{
    int gid = blockIdx.x * blockDim.x + threadIdx.x;
    int p = gid >> 5;
    int c = (gid & 31) * 4;
    if (p >= M_TOT) return;
    float4 v = *(const float4*)(prev + (size_t)p*C + c);
    __bf16* o = cat + (size_t)p*K0P + 85 + c;
    o[0] = (__bf16)v.x; o[1] = (__bf16)v.y;
    o[2] = (__bf16)v.z; o[3] = (__bf16)v.w;
}

// ---------------------------------------------------------------------------
// Kernel 4: correlation. One wave per pixel; each lane owns displacements
// {lane, lane+32, lane+64} (<81). 128-ch fp32 dot via float4, mean -> bf16
// into concat channels [4..85). Padded buffer => no bounds checks.
// ---------------------------------------------------------------------------
__global__ void corr_kernel(const float* __restrict__ prev,
                            const float* __restrict__ pad,
                            __bf16* __restrict__ cat)
{
    int tid  = threadIdx.x;
    int p    = blockIdx.x * 8 + (tid >> 5);
    int lane = tid & 31;
    if (p >= M_TOT) return;
    int b  = p / (H*W);
    int rm = p - b*(H*W);
    int y  = rm / W;
    int x  = rm - y*W;
    const float* a = prev + (size_t)p*C;
    for (int rep = 0; rep < 3; ++rep) {
        int d = lane + rep*32;
        if (d < 81) {
            int dy = d / 9;
            int dx = d - dy*9;
            const float* bb = pad + (((size_t)(b*HP + (y+dy))*WP + (x+dx))*C);
            float s = 0.f;
            for (int c0 = 0; c0 < C; c0 += 4) {
                float4 av = *(const float4*)(a  + c0);
                float4 bv = *(const float4*)(bb + c0);
                s += av.x*bv.x + av.y*bv.y + av.z*bv.z + av.w*bv.w;
            }
            cat[(size_t)p*K0P + 4 + d] = (__bf16)(s * (1.0f/128.0f));
        }
    }
}

// ---------------------------------------------------------------------------
// Kernel 5: depthwise 3x3 'SAME', bf16 in -> bf16 out, fp32 accumulate.
// Thread per (pixel, 4 channels). Channels >= Creal produce 0 (zero weights).
// ---------------------------------------------------------------------------
__global__ void dw_kernel(const __bf16* __restrict__ in,
                          const float* __restrict__ w33,  // [(kh*3+kw)*Creal + c]
                          __bf16* __restrict__ out,
                          int Kp, int Creal)
{
    int gid = blockIdx.x * blockDim.x + threadIdx.x;
    int groups = Kp >> 2;
    int p = gid / groups;
    int c = (gid - p*groups) * 4;
    if (p >= M_TOT) return;
    int b  = p / (H*W);
    int rm = p - b*(H*W);
    int y  = rm / W;
    int x  = rm - y*W;

    float a0 = 0.f, a1 = 0.f, a2 = 0.f, a3 = 0.f;
    for (int kh = 0; kh < 3; ++kh) {
        int yy = y + kh - 1;
        if ((unsigned)yy >= (unsigned)H) continue;
        for (int kw = 0; kw < 3; ++kw) {
            int xx = x + kw - 1;
            if ((unsigned)xx >= (unsigned)W) continue;
            const __bf16* ip = in + ((size_t)(b*H + yy)*W + xx)*Kp + c;
            v4bf iv = *(const v4bf*)ip;
            const float* wp = w33 + (size_t)(kh*3 + kw)*Creal + c;
            float w0 = (c+0 < Creal) ? wp[0] : 0.f;
            float w1 = (c+1 < Creal) ? wp[1] : 0.f;
            float w2 = (c+2 < Creal) ? wp[2] : 0.f;
            float w3 = (c+3 < Creal) ? wp[3] : 0.f;
            a0 += (float)iv.x * w0;
            a1 += (float)iv.y * w1;
            a2 += (float)iv.z * w2;
            a3 += (float)iv.w * w3;
        }
    }
    __bf16* op = out + (size_t)p*Kp + c;
    op[0] = (__bf16)a0; op[1] = (__bf16)a1;
    op[2] = (__bf16)a2; op[3] = (__bf16)a3;
}

// ---------------------------------------------------------------------------
// Kernel 6: pointwise 1x1 conv as WMMA bf16 GEMM.
//   out[M,128] = relu( act[M,Kp] x wts[Kreal,128] + bias )
// Block: 256 threads = 8 waves; tile M=128, N=128.
// Data movement (gfx1250 paths):
//   * A tile (contiguous 128*Kp bf16): TENSOR_LOAD_TO_LDS (TDM, 1-D tile,
//     data_size=2B, tile_dim0=128*Kp, lds_addr=0) -> s_wait_tensorcnt.
//   * Weights (contiguous Kreal*128 fp32): GLOBAL_LOAD_ASYNC_TO_LDS_B128 into
//     scratch sW -> s_wait_asynccnt, then in-LDS transpose+cvt into sB [N][K].
// LDS: sA [128][Kp] bf16 | sB [128][Kp] bf16 | sW [Kreal][128] fp32
// A fragment (16-bit 16x32): lane<16: M=lane, K {kb..kb+7, kb+16..kb+23}
//                            lane>=16: M=lane-16, K {+8, +24}   (ISA 7.12.2)
// B fragment (32x16):        lane<16: N=lane, K {kb..kb+7, kb+16..kb+23}
//                            lane>=16: N=lane-16, K {+8, +24}
// ---------------------------------------------------------------------------
__global__ void pw_gemm_kernel(const __bf16* __restrict__ act,
                               const float* __restrict__ wts,   // [K][128] fp32
                               const float* __restrict__ bias,  // [128]
                               __bf16* __restrict__ outb,       // [M][128] bf16
                               float* __restrict__ outf,        // optional fp32 copy
                               int Kp, int Kreal)
{
    extern __shared__ char smem_raw[];
    __bf16* sA = (__bf16*)smem_raw;                        // 128*Kp bf16 @ LDS 0
    __bf16* sB = (__bf16*)(smem_raw + (size_t)128*Kp*2);   // 128*Kp bf16
    float*  sW = (float*) (smem_raw + (size_t)128*Kp*4);   // Kreal*128 fp32

    const int tid     = threadIdx.x;
    const int rowBase = blockIdx.x * 128;

    // --- A tile via Tensor Data Mover: one 1-D tile descriptor (D#) ---
    if (tid == 0) {
        unsigned long long ga = (unsigned long long)(const void*)(act + (size_t)rowBase*Kp);
        unsigned tile0 = (unsigned)(128*Kp);               // elements of 2 bytes
        u32x4 g0;
        g0[0] = 1u;                                        // count=1, user D#
        g0[1] = 0u;                                        // lds_addr = 0 (sA)
        g0[2] = (unsigned)(ga & 0xFFFFFFFFu);              // global_addr[31:0]
        g0[3] = (unsigned)((ga >> 32) & 0x01FFFFFFu) | 0x80000000u; // [56:32]|type=2
        i32x8 g1;
        g1[0] = (int)(1u << 16);                           // data_size=1 (2B)
        g1[1] = (int)((tile0 & 0xFFFFu) << 16);            // tensor_dim0[15:0]
        g1[2] = (int)((tile0 >> 16) & 0xFFFFu);            // tensor_dim0[31:16]
        g1[3] = (int)(tile0 << 16);                        // tile_dim0
        g1[4] = 0;                                         // tile_dim1/2 = 0
        g1[5] = (int)tile0;                                // tensor_dim0_stride lo
        g1[6] = 0;
        g1[7] = 0;
        i32x4 g2 = (i32x4){0,0,0,0};
        i32x4 g3 = (i32x4){0,0,0,0};
        i32x8 g4 = (i32x8){0,0,0,0,0,0,0,0};
        __builtin_amdgcn_tensor_load_to_lds(g0, g1, g2, g3, g4, 0);
    }

    // --- raw fp32 weights via async global->LDS (contiguous Kreal*128*4 B) ---
    {
        const char* gW = (const char*)wts;
        const unsigned swOff = (unsigned)(128*Kp*4);       // sW LDS byte offset
        const int nchW = (Kreal*128*4) >> 4;               // 16-byte chunks
        for (int i = tid; i < nchW; i += 256) {
            unsigned goff = (unsigned)(i << 4);
            unsigned loff = swOff + goff;
            asm volatile("global_load_async_to_lds_b128 %0, %1, %2"
                         :: "v"(loff), "v"(goff), "s"(gW)
                         : "memory");
        }
        if (rowBase + 128 < M_TOT)   // warm L2 for the next M-tile
            __builtin_prefetch(act + (size_t)(rowBase + 128)*Kp + tid*64, 0, 1);
    }
    asm volatile("s_wait_asynccnt 0" ::: "memory");   // own async chunks done
    __builtin_amdgcn_s_wait_tensorcnt(0);             // TDM tile done (wave 0)
    __syncthreads();                                  // all staging visible

    // --- in-LDS transpose + fp32->bf16 convert: sW [K][128] -> sB [N][K] ---
    {
        int tot = 128 * Kp;
        for (int i = tid; i < tot; i += 256) {
            int n = i / Kp;
            int k = i - n*Kp;
            float wv = (k < Kreal) ? sW[(size_t)k*128 + n] : 0.f;
            sB[(size_t)n*Kp + k] = (__bf16)wv;
        }
    }
    __syncthreads();

    const int wid  = tid >> 5;
    const int lane = tid & 31;
    const int half = lane >> 4;
    const int m    = lane & 15;

    v8f acc[8];
#pragma unroll
    for (int i = 0; i < 8; ++i)
        acc[i] = (v8f){0.f,0.f,0.f,0.f,0.f,0.f,0.f,0.f};

    const char* aRow = (const char*)(sA + (size_t)(wid*16 + m)*Kp);
    const int ksteps = Kp >> 5;
    for (int ks = 0; ks < ksteps; ++ks) {
        const int kb = ks * 32;
        FragU fa;
        {
            int aoff = (kb + half*8) * 2;
            fa.u2[0] = *(const v4u*)(aRow + aoff);        // K kb..+7   (or +8)
            fa.u2[1] = *(const v4u*)(aRow + aoff + 32);   // K kb+16..+23 (or +24)
        }
        FragU fb[8];
        const int boff = (kb + half*8) * 2;
#pragma unroll
        for (int nt = 0; nt < 8; ++nt) {
            const char* bRow = (const char*)(sB + (size_t)(nt*16 + m)*Kp);
            fb[nt].u2[0] = *(const v4u*)(bRow + boff);
            fb[nt].u2[1] = *(const v4u*)(bRow + boff + 32);
        }
#pragma unroll
        for (int nt = 0; nt < 8; ++nt)
            acc[nt] = __builtin_amdgcn_wmma_f32_16x16x32_bf16(
                false, fa.v, false, fb[nt].v, (short)0, acc[nt], false, false);
    }

    // C/D layout: VGPR j -> (M = j + half*8, N = nt*16 + m). Bias + ReLU fused.
#pragma unroll
    for (int nt = 0; nt < 8; ++nt) {
        int col = nt*16 + m;
        float bv = bias[col];
#pragma unroll
        for (int j = 0; j < 8; ++j) {
            int row = rowBase + wid*16 + j + half*8;
            float v = fmaxf(acc[nt][j] + bv, 0.f);
            outb[(size_t)row*128 + col] = (__bf16)v;
            if (outf) outf[(size_t)row*128 + col] = v;
        }
    }
}

// ---------------------------------------------------------------------------
// Kernel 7: tiny heads (N=2 flow / N=1 mask): pointwise over 128 bf16 channels
// ---------------------------------------------------------------------------
__global__ void head_kernel(const __bf16* __restrict__ dwb,  // [M][128] bf16
                            const float* __restrict__ pw,    // [128][Nh]
                            const float* __restrict__ hb,    // [Nh]
                            int Nh, float* __restrict__ outp)
{
    int p = blockIdx.x * blockDim.x + threadIdx.x;
    if (p >= M_TOT) return;
    float a0 = 0.f, a1 = 0.f;
    const __bf16* r = dwb + (size_t)p*NF;
    for (int c = 0; c < NF; ++c) {
        float v = (float)r[c];
        a0 += v * pw[c*Nh + 0];
        if (Nh > 1) a1 += v * pw[c*Nh + 1];
    }
    outp[(size_t)p*Nh + 0] = a0 + hb[0];
    if (Nh > 1) outp[(size_t)p*Nh + 1] = a1 + hb[1];
}

// ---------------------------------------------------------------------------
// launcher
// ---------------------------------------------------------------------------
extern "C" void kernel_launch(void* const* d_in, const int* in_sizes, int n_in,
                              void* d_out, int out_size, void* d_ws, size_t ws_size,
                              hipStream_t stream)
{
    const float* sf      = (const float*)d_in[0];
    const float* sfl     = (const float*)d_in[1];
    const float* sm      = (const float*)d_in[2];
    const float* prev    = (const float*)d_in[3];
    const float* cur     = (const float*)d_in[4];
    const float* upf_k   = (const float*)d_in[5];
    const float* upf_b   = (const float*)d_in[6];
    const float* upfl_k  = (const float*)d_in[7];
    const float* upfl_b  = (const float*)d_in[8];
    const float* flow_dw = (const float*)d_in[9];
    const float* flow_pw = (const float*)d_in[10];
    const float* flow_b  = (const float*)d_in[11];
    const float* mask_dw = (const float*)d_in[12];
    const float* mask_pw = (const float*)d_in[13];
    const float* mask_b  = (const float*)d_in[14];
    const float* dws[4], *pws[4], *bs[4];
    for (int i = 0; i < 4; ++i) {
        dws[i] = (const float*)d_in[15 + i];
        pws[i] = (const float*)d_in[19 + i];
        bs[i]  = (const float*)d_in[23 + i];
    }

    // workspace layout (pad region is reused as depthwise scratch after corr)
    const size_t PAD_BYTES  = (size_t)BATCH*HP*WP*C*4;      //  75,759,616
    const size_t CAT_BYTES  = (size_t)M_TOT*K0P*2;          //  58,720,256
    const size_t GATE_BYTES = (size_t)M_TOT*4;              //     524,288
    const size_t ACT_BYTES  = (size_t)M_TOT*NF*2;           //  33,554,432
    char* ws = (char*)d_ws;
    float*  pad   = (float*)(ws);
    __bf16* dwbuf = (__bf16*)(ws);                           // aliases pad
    __bf16* cat   = (__bf16*)(ws + PAD_BYTES);
    float*  gate  = (float*)(ws + PAD_BYTES + CAT_BYTES);
    __bf16* actA  = (__bf16*)(ws + PAD_BYTES + CAT_BYTES + GATE_BYTES);
    __bf16* actB  = (__bf16*)((char*)actA + ACT_BYTES);
    (void)ws_size; // requires ~202 MB

    float* out_x      = (float*)d_out;                        // [M][128]
    float* out_flow   = out_x + (size_t)M_TOT*NF;             // [M][2]
    float* out_mask   = out_flow + (size_t)M_TOT*2;           // [M][1]
    float* out_upflow = out_mask + (size_t)M_TOT;             // [M][2]

    (void)hipMemsetAsync(pad, 0, PAD_BYTES, stream);   // zero pad borders
    (void)hipMemsetAsync(cat, 0, CAT_BYTES, stream);   // zero K-pad ch 213..223

    upsample_kernel<<<M_TOT/256, 256, 0, stream>>>(
        sf, sfl, sm, upf_k, upf_b, upfl_k, upfl_b, cat, out_upflow, gate);
    warp_kernel<<<(M_TOT*32)/256, 256, 0, stream>>>(cur, out_upflow, gate, pad);
    prevcat_kernel<<<(M_TOT*32)/256, 256, 0, stream>>>(prev, cat);
    corr_kernel<<<M_TOT/8, 256, 0, stream>>>(prev, pad, cat);

    (void)hipFuncSetAttribute(reinterpret_cast<const void*>(pw_gemm_kernel),
                              hipFuncAttributeMaxDynamicSharedMemorySize, 1024*K0P);

    // block 0: K = 213 (padded 224); dynamic LDS = sA+sB+sW = 1024*Kp bytes
    dw_kernel<<<(M_TOT*(K0P/4))/256, 256, 0, stream>>>(cat, dws[0], dwbuf, K0P, CIN);
    pw_gemm_kernel<<<M_TOT/128, 256, 1024*K0P, stream>>>(
        dwbuf, pws[0], bs[0], actA, nullptr, K0P, CIN);

    // blocks 1..3: K = 128
    __bf16* pin = actA; __bf16* pout = actB;
    for (int i = 1; i < 4; ++i) {
        dw_kernel<<<(M_TOT*(NF/4))/256, 256, 0, stream>>>(pin, dws[i], dwbuf, NF, NF);
        pw_gemm_kernel<<<M_TOT/128, 256, 1024*NF, stream>>>(
            dwbuf, pws[i], bs[i], pout, (i == 3) ? out_x : nullptr, NF, NF);
        __bf16* t = pin; pin = pout; pout = t;
    }

    // heads on final x (bf16 in `pin`)
    dw_kernel<<<(M_TOT*(NF/4))/256, 256, 0, stream>>>(pin, flow_dw, dwbuf, NF, NF);
    head_kernel<<<M_TOT/256, 256, 0, stream>>>(dwbuf, flow_pw, flow_b, 2, out_flow);
    dw_kernel<<<(M_TOT*(NF/4))/256, 256, 0, stream>>>(pin, mask_dw, dwbuf, NF, NF);
    head_kernel<<<M_TOT/256, 256, 0, stream>>>(dwbuf, mask_pw, mask_b, 1, out_mask);
}
